// MultiHeadAttention_49082886259242
// MI455X (gfx1250) — matrix operand order repaired and checked by reading, not compiled, not run on hardware
//
#include <hip/hip_runtime.h>

typedef __attribute__((ext_vector_type(16))) __bf16 v16bf;
typedef __attribute__((ext_vector_type(8)))  __bf16 v8bf;
typedef __attribute__((ext_vector_type(4)))  __bf16 v4bf;
typedef __attribute__((ext_vector_type(8)))  float  v8f;
typedef __attribute__((ext_vector_type(4)))  float  v4f;
typedef __attribute__((ext_vector_type(4)))  unsigned int v4u;
typedef __attribute__((ext_vector_type(8)))  int    v8i;
typedef __attribute__((ext_vector_type(4)))  int    v4i;

#define BATCH   4
#define SEQ     2048
#define NHEADS  16
#define DMODEL  1024
#define DHEAD   64
#define BSROWS  (BATCH*SEQ)      /* 8192 */

#if defined(__has_builtin)
#if __has_builtin(__builtin_amdgcn_tensor_load_to_lds) && \
    __has_builtin(__builtin_amdgcn_s_wait_tensorcnt)
#define USE_TDM 1
#endif
#endif
#ifndef USE_TDM
#define USE_TDM 0
#endif

// ---------------------------------------------------------------------------
// Elementwise f32 -> bf16 (4 elems / thread)
// ---------------------------------------------------------------------------
__global__ __launch_bounds__(256) void f32_to_bf16_kernel(
    const float* __restrict__ in, __bf16* __restrict__ out, long long n4)
{
    long long i = (long long)blockIdx.x * 256 + threadIdx.x;
    if (i >= n4) return;
    v4f a = *(const v4f*)(in + i * 4);
    v4bf o;
#pragma unroll
    for (int j = 0; j < 4; ++j) o[j] = (__bf16)a[j];
    *(v4bf*)(out + i * 4) = o;
}

// ---------------------------------------------------------------------------
// Weight transpose + convert: in [K=1024][N=1024] f32 -> out [N][K] bf16
// ---------------------------------------------------------------------------
__global__ __launch_bounds__(256) void transpose_w_kernel(
    const float* __restrict__ in, __bf16* __restrict__ out)
{
    int i = blockIdx.x * 256 + threadIdx.x;          // 0 .. 262143
    int k4 = (i & 255) * 4;
    int n  = i >> 8;
    v4bf o;
#pragma unroll
    for (int j = 0; j < 4; ++j)
        o[j] = (__bf16)in[(long long)(k4 + j) * DMODEL + n];
    *(v4bf*)(out + (long long)n * DMODEL + k4) = o;
}

// ---------------------------------------------------------------------------
// Pack projection f32 [B][S][H*64] -> bf16 [B][H][S][64], with scale
// ---------------------------------------------------------------------------
__global__ __launch_bounds__(256) void pack_heads_kernel(
    const float* __restrict__ in, __bf16* __restrict__ out, float scale)
{
    int i = blockIdx.x * 256 + threadIdx.x;          // 4 elems each
    if (i >= (BSROWS * DMODEL) / 4) return;
    int d4 = i & 15;
    int h  = (i >> 4) & 15;
    int s  = (i >> 8) & (SEQ - 1);
    int b  = i >> 19;
    const float* src = in + (((long long)b * SEQ + s) * DMODEL) + h * 64 + d4 * 4;
    v4f a = *(const v4f*)src;
    v4bf o;
#pragma unroll
    for (int j = 0; j < 4; ++j) o[j] = (__bf16)(a[j] * scale);
    __bf16* dst = out + ((((long long)b * NHEADS + h) * SEQ + s) * 64) + d4 * 4;
    *(v4bf*)dst = o;
}

// ---------------------------------------------------------------------------
// Pack v projection f32 [B][S][H*64] -> bf16 vT [B][H][64][S]
// ---------------------------------------------------------------------------
__global__ __launch_bounds__(256) void pack_vT_kernel(
    const float* __restrict__ in, __bf16* __restrict__ out)
{
    int i = blockIdx.x * 256 + threadIdx.x;          // 4 output elems each
    if (i >= (BSROWS * DMODEL) / 4) return;
    int s4 = i & 511;              // S/4
    int d  = (i >> 9) & 63;
    int h  = (i >> 15) & 15;
    int b  = i >> 19;
    v4bf o;
#pragma unroll
    for (int j = 0; j < 4; ++j) {
        float f = in[(((long long)b * SEQ + s4 * 4 + j) * DMODEL) + h * 64 + d];
        o[j] = (__bf16)f;
    }
    __bf16* dst = out + ((((long long)b * NHEADS + h) * 64 + d) * SEQ) + s4 * 4;
    *(v4bf*)dst = o;
}

#if USE_TDM
// ---------------------------------------------------------------------------
// Issue a TDM 2D tile load: rows x 32 bf16 tile, row stride = strideElems,
// into LDS at ldsOff with 16B pad after each 64B row (-> pitch 40 elements).
// D# layout per CDNA5 ISA.
// This toolchain's builtin is the 6-arg form:
//   (uint32x4 g0, int32x8 g1, int32x4 g2, int32x4 g3, int32x8 gx, i32 cpol)
// groups 2/3 (and the trailing group) are zero for a 2D tensor.
// ---------------------------------------------------------------------------
__device__ __forceinline__ void tdm_load_tile(
    const __bf16* gaddr, unsigned ldsOff, int rows, long long strideElems)
{
    unsigned long long ga = (unsigned long long)gaddr;
    v4u g0;
    g0[0] = 1u;                                          // count=1 (valid user D#)
    g0[1] = ldsOff;                                      // lds_addr (bytes)
    g0[2] = (unsigned)(ga & 0xFFFFFFFFu);                // global_addr[31:0]
    g0[3] = (unsigned)((ga >> 32) & 0x01FFFFFFu)         // global_addr[56:32]
          | (2u << 30);                                  // type = 2 ("image")
    v8i g1;
    g1[0] = (1 << 16)        // data_size = 1 -> 2 bytes
          | (1 << 20)        // pad_enable
          | (3 << 22)        // pad_interval code 3 -> every 16 DWORDs (64B)
          | (3 << 25);       // pad_amount  code 3 -> 4 DWORDs (16B)
    g1[1] = (32 & 0xFFFF) << 16;                 // tensor_dim0 = 32 (lo16)
    g1[2] = ((rows & 0xFFFF) << 16);             // tensor_dim0 hi=0 | tensor_dim1 lo16
    g1[3] = (32 << 16);                          // tensor_dim1 hi=0 | tile_dim0 = 32
    g1[4] = rows & 0xFFFF;                       // tile_dim1 | tile_dim2 = 0
    g1[5] = (int)(strideElems & 0xFFFFFFFFll);   // tensor_dim0_stride lo32 (elements)
    g1[6] = (int)((strideElems >> 32) & 0xFFFF); // stride hi16 | dim1_stride lo = 0
    g1[7] = 0;
    v4i gz4 = {0, 0, 0, 0};
    v8i gz8 = {0, 0, 0, 0, 0, 0, 0, 0};
    __builtin_amdgcn_tensor_load_to_lds(g0, g1, gz4, gz4, gz8, 0);
}

__device__ __forceinline__ unsigned lds_off(const void* p)
{
    return (unsigned)(unsigned long long)p;              // low 32 bits = LDS byte addr
}
#endif

// ---------------------------------------------------------------------------
// Simple NT GEMM: C = A * B^T + bias.  A [M][K] bf16, B [N][K] bf16, C f32.
// Tile 128x64, BK=32, 256 threads = 8 waves (wave32), wave = 32x32 block.
// TDM path: double-buffered Tensor-Data-Mover DMA of the A/B tiles into LDS,
// pipelined on TENSORcnt (issue next tile, wait <=2 so current tile is done).
// ---------------------------------------------------------------------------
__global__ __launch_bounds__(256) void gemm_nt_bias_kernel(
    const __bf16* __restrict__ A, const __bf16* __restrict__ B,
    float* __restrict__ C, const float* __restrict__ bias, int K)
{
    constexpr int LDSW = 40;                 // 80B pitch -> conflict-free b128 reads
    __shared__ __align__(16) __bf16 As[2][128 * LDSW];
    __shared__ __align__(16) __bf16 Bs[2][64 * LDSW];

    const int t    = threadIdx.x;
    const int lane = t & 31;
    const int wid  = t >> 5;
    const int hi16 = lane >> 4;
    const int l15  = lane & 15;
    const int waveM = (wid >> 1) * 32;
    const int waveN = (wid & 1) * 32;
    const int mBase = blockIdx.y * 128;
    const int nBase = blockIdx.x * 64;

    v8f acc[2][2];
#pragma unroll
    for (int i = 0; i < 2; ++i)
#pragma unroll
        for (int j = 0; j < 2; ++j)
#pragma unroll
            for (int r = 0; r < 8; ++r) acc[i][j][r] = 0.0f;

#if USE_TDM
    const __bf16* Abase = A + (long long)mBase * K;
    const __bf16* Bbase = B + (long long)nBase * K;
    const int steps = K >> 5;
    if (wid == 0) {
        tdm_load_tile(Abase, lds_off(&As[0][0]), 128, K);
        tdm_load_tile(Bbase, lds_off(&Bs[0][0]), 64, K);
    }
    for (int i = 0; i < steps; ++i) {
        const int cur = i & 1;
        if (wid == 0) {
            if (i + 1 < steps) {
                tdm_load_tile(Abase + (i + 1) * 32, lds_off(&As[cur ^ 1][0]), 128, K);
                tdm_load_tile(Bbase + (i + 1) * 32, lds_off(&Bs[cur ^ 1][0]), 64, K);
                __builtin_amdgcn_s_wait_tensorcnt(2);   // tile i complete (in-order)
            } else {
                __builtin_amdgcn_s_wait_tensorcnt(0);
            }
        }
        __syncthreads();                                 // tile i visible to all waves
        const __bf16* Asb = &As[cur][0];
        const __bf16* Bsb = &Bs[cur][0];
#else
    for (int k0 = 0; k0 < K; k0 += 32) {
        const int srow = t >> 2;                 // 0..63
        const int scol = (t & 3) * 8;            // 0,8,16,24
#pragma unroll
        for (int p = 0; p < 2; ++p) {
            int r = srow + p * 64;
            v8bf v = *(const v8bf*)(A + (long long)(mBase + r) * K + (k0 + scol));
            *(v8bf*)(&As[0][r * LDSW + scol]) = v;
        }
        {
            v8bf v = *(const v8bf*)(B + (long long)(nBase + srow) * K + (k0 + scol));
            *(v8bf*)(&Bs[0][srow * LDSW + scol]) = v;
        }
        __syncthreads();
        const __bf16* Asb = &As[0][0];
        const __bf16* Bsb = &Bs[0][0];
#endif

        v16bf afr[2], bfr[2];
#pragma unroll
        for (int mt = 0; mt < 2; ++mt) {
            const __bf16* base = &Asb[(waveM + mt * 16 + l15) * LDSW];
            union { v16bf v; v8bf x[2]; } u;
            u.x[0] = *(const v8bf*)(base + hi16 * 8);
            u.x[1] = *(const v8bf*)(base + 16 + hi16 * 8);
            afr[mt] = u.v;
        }
#pragma unroll
        for (int nt = 0; nt < 2; ++nt) {
            const __bf16* base = &Bsb[(waveN + nt * 16 + l15) * LDSW + hi16 * 8];
            union { v16bf v; v8bf x[2]; } u;
            u.x[0] = *(const v8bf*)(base);
            u.x[1] = *(const v8bf*)(base + 16);
            bfr[nt] = u.v;
        }

#pragma unroll
        for (int mt = 0; mt < 2; ++mt)
#pragma unroll
            for (int nt = 0; nt < 2; ++nt)
                acc[mt][nt] = __builtin_amdgcn_wmma_f32_16x16x32_bf16(
                    false, afr[mt], false, bfr[nt], (short)0, acc[mt][nt], false, false);

        __syncthreads();
    }

#pragma unroll
    for (int mt = 0; mt < 2; ++mt)
#pragma unroll
        for (int nt = 0; nt < 2; ++nt) {
            int n = nBase + waveN + nt * 16 + l15;
            float bvv = bias[n];
#pragma unroll
            for (int r = 0; r < 8; ++r) {
                int m = mBase + waveM + mt * 16 + hi16 * 8 + r;
                C[(long long)m * DMODEL + n] = acc[mt][nt][r] + bvv;
            }
        }
}

// ---------------------------------------------------------------------------
// Fused scores + softmax + context.
// Grid: (SEQ/16, B*H), block 512 = 16 waves.
// Each block owns a 16-row x 2048-col score panel of one head, kept in
// registers (wave w holds columns [w*128, w*128+128) = 8 WMMA accumulators).
// ---------------------------------------------------------------------------
__global__ __launch_bounds__(512) void fused_attn_kernel(
    const __bf16* __restrict__ qh,   // [B*H][S][64]
    const __bf16* __restrict__ kh,   // [B*H][S][64]
    const __bf16* __restrict__ vT,   // [B*H][64][S]
    float* __restrict__ attnF,       // [(h*B+b)][S][S]
    __bf16* __restrict__ ctx)        // [B][S][H*64]
{
    constexpr int APW = 1032;        // half-panel pitch: 2064B = 4 banks mod 64
    __shared__ __align__(16) __bf16 apan[16 * APW];   // 33 KB
    __shared__ float smax[16 * 16];
    __shared__ float ssum[16 * 16];
    __shared__ float cctx[16 * 64];

    const int t    = threadIdx.x;
    const int lane = t & 31;
    const int wid  = t >> 5;          // 0..15
    const int hi16 = lane >> 4;
    const int l15  = lane & 15;

    const int z  = blockIdx.y;        // b*H + h
    const int b  = z >> 4;
    const int h  = z & 15;
    const int m0 = blockIdx.x * 16;   // row panel base

    const long long zoff = (long long)z * SEQ * DHEAD;
    const __bf16* qbase = qh + zoff + (long long)m0 * DHEAD;
    const __bf16* kbase = kh + zoff;
    const __bf16* vbase = vT + (long long)z * DHEAD * SEQ;

    // ---------------- Phase 1: scores panel in registers ----------------
    v16bf af[2];
#pragma unroll
    for (int kk = 0; kk < 2; ++kk) {
        const __bf16* ap = qbase + (long long)l15 * DHEAD + kk * 32 + hi16 * 8;
        union { v16bf v; v8bf x[2]; } u;
        u.x[0] = *(const v8bf*)(ap);
        u.x[1] = *(const v8bf*)(ap + 16);
        af[kk] = u.v;
    }

    const int col0 = wid * 128;
    v8f acc[8];
#pragma unroll
    for (int nt = 0; nt < 8; ++nt)
#pragma unroll
        for (int r = 0; r < 8; ++r) acc[nt][r] = 0.0f;

#pragma unroll
    for (int nt = 0; nt < 8; ++nt) {
        const __bf16* krow = kbase + (long long)(col0 + nt * 16 + l15) * DHEAD;
#pragma unroll
        for (int kk = 0; kk < 2; ++kk) {
            union { v16bf v; v8bf x[2]; } u;
            u.x[0] = *(const v8bf*)(krow + kk * 32 + hi16 * 8);
            u.x[1] = *(const v8bf*)(krow + kk * 32 + hi16 * 8 + 16);
            acc[nt] = __builtin_amdgcn_wmma_f32_16x16x32_bf16(
                false, af[kk], false, u.v, (short)0, acc[nt], false, false);
        }
    }

    // ---------------- Phase 2: softmax over each of the 16 rows ----------------
    float rmax[8];
#pragma unroll
    for (int r = 0; r < 8; ++r) {
        float m = acc[0][r];
#pragma unroll
        for (int nt = 1; nt < 8; ++nt) m = fmaxf(m, acc[nt][r]);
#pragma unroll
        for (int msk = 1; msk < 16; msk <<= 1) m = fmaxf(m, __shfl_xor(m, msk, 32));
        rmax[r] = m;
    }
    if (l15 == 0) {
#pragma unroll
        for (int r = 0; r < 8; ++r) smax[wid * 16 + hi16 * 8 + r] = rmax[r];
    }
    __syncthreads();
#pragma unroll
    for (int r = 0; r < 8; ++r) {
        float m = smax[hi16 * 8 + r];
        for (int w2 = 1; w2 < 16; ++w2) m = fmaxf(m, smax[w2 * 16 + hi16 * 8 + r]);
        rmax[r] = m;
    }

    float rsum[8];
#pragma unroll
    for (int r = 0; r < 8; ++r) rsum[r] = 0.0f;
#pragma unroll
    for (int nt = 0; nt < 8; ++nt)
#pragma unroll
        for (int r = 0; r < 8; ++r) {
            float e = __expf(acc[nt][r] - rmax[r]);
            acc[nt][r] = e;
            rsum[r] += e;
        }
#pragma unroll
    for (int r = 0; r < 8; ++r) {
        float s = rsum[r];
#pragma unroll
        for (int msk = 1; msk < 16; msk <<= 1) s += __shfl_xor(s, msk, 32);
        rsum[r] = s;
    }
    if (l15 == 0) {
#pragma unroll
        for (int r = 0; r < 8; ++r) ssum[wid * 16 + hi16 * 8 + r] = rsum[r];
    }
    __syncthreads();
#pragma unroll
    for (int r = 0; r < 8; ++r) {
        float s = 0.0f;
        for (int w2 = 0; w2 < 16; ++w2) s += ssum[w2 * 16 + hi16 * 8 + r];
        float inv = 1.0f / s;
#pragma unroll
        for (int nt = 0; nt < 8; ++nt) acc[nt][r] *= inv;
    }

    // ---------------- Phase 3: write normalized attn (f32) ----------------
    float* abase = attnF + ((long long)(h * BATCH + b) * SEQ + m0) * SEQ;
#pragma unroll
    for (int nt = 0; nt < 8; ++nt) {
        int n = col0 + nt * 16 + l15;
#pragma unroll
        for (int r = 0; r < 8; ++r)
            abase[(long long)(hi16 * 8 + r) * SEQ + n] = acc[nt][r];
    }

    // convert own panel slice to bf16; acc dies here (frees 64 VGPRs)
    v8bf abf[8];
#pragma unroll
    for (int nt = 0; nt < 8; ++nt)
#pragma unroll
        for (int r = 0; r < 8; ++r) abf[nt][r] = (__bf16)acc[nt][r];

    // ---------------- Phase 4: context = attn . v ----------------
    for (int e = t; e < 1024; e += 512) cctx[e] = 0.0f;

    v8f cacc[4];
#pragma unroll
    for (int nt = 0; nt < 4; ++nt)
#pragma unroll
        for (int r = 0; r < 8; ++r) cacc[nt][r] = 0.0f;

    for (int p = 0; p < 2; ++p) {
        __syncthreads();   // protects apan reuse (and cctx zeroing on p=0)
        if ((wid >> 3) == p) {                 // waves owning this half-panel
            int lcol0 = (wid & 7) * 128;
#pragma unroll
            for (int nt = 0; nt < 8; ++nt) {
                int lc = lcol0 + nt * 16 + l15;
#pragma unroll
                for (int r = 0; r < 8; ++r)
                    apan[(hi16 * 8 + r) * APW + lc] = abf[nt][r];
            }
        }
        __syncthreads();
        // wave wid contracts local k range [wid*64, wid*64+64)
#pragma unroll
        for (int kk = 0; kk < 2; ++kk) {
            int kl = wid * 64 + kk * 32;
            const __bf16* ap = &apan[l15 * APW + kl + hi16 * 8];
            union { v16bf v; v8bf x[2]; } ua;
            ua.x[0] = *(const v8bf*)(ap);
            ua.x[1] = *(const v8bf*)(ap + 16);
            int gk = p * 1024 + kl;
#pragma unroll
            for (int nt = 0; nt < 4; ++nt) {
                const __bf16* vrow = vbase + (long long)(nt * 16 + l15) * SEQ + gk;
                union { v16bf v; v8bf x[2]; } ub;
                ub.x[0] = *(const v8bf*)(vrow + hi16 * 8);
                ub.x[1] = *(const v8bf*)(vrow + hi16 * 8 + 16);
                cacc[nt] = __builtin_amdgcn_wmma_f32_16x16x32_bf16(
                    false, ua.v, false, ub.v, (short)0, cacc[nt], false, false);
            }
        }
    }

    // deterministic serialized cross-wave reduction into cctx
    __syncthreads();
    for (int w2 = 0; w2 < 16; ++w2) {
        if (wid == w2) {
#pragma unroll
            for (int nt = 0; nt < 4; ++nt)
#pragma unroll
                for (int r = 0; r < 8; ++r)
                    cctx[(hi16 * 8 + r) * 64 + nt * 16 + l15] += cacc[nt][r];
        }
        __syncthreads();
    }

    for (int e = t; e < 1024; e += 512) {
        int mr = e >> 6, d = e & 63;
        ctx[(long long)(b * SEQ + m0 + mr) * DMODEL + h * 64 + d] = (__bf16)cctx[e];
    }
}

// ---------------------------------------------------------------------------
// Host-side orchestration
// ---------------------------------------------------------------------------
extern "C" void kernel_launch(void* const* d_in, const int* in_sizes, int n_in,
                              void* d_out, int out_size, void* d_ws, size_t ws_size,
                              hipStream_t stream)
{
    (void)in_sizes; (void)n_in; (void)out_size; (void)ws_size;
    const float* Q  = (const float*)d_in[0];
    const float* K  = (const float*)d_in[1];
    const float* V  = (const float*)d_in[2];
    const float* Wq = (const float*)d_in[3];
    const float* bq = (const float*)d_in[4];
    const float* Wk = (const float*)d_in[5];
    const float* bk = (const float*)d_in[6];
    const float* Wv = (const float*)d_in[7];
    const float* bv = (const float*)d_in[8];
    const float* Wo = (const float*)d_in[9];
    const float* bo = (const float*)d_in[10];

    float* outF  = (float*)d_out;                                 // [8192][1024]
    float* attnF = outF + (long long)BSROWS * DMODEL;             // [64][2048][2048]

    char* ws = (char*)d_ws;
    __bf16* Xq   = (__bf16*)(ws + 0LL);
    __bf16* Xk   = (__bf16*)(ws + 16777216LL);
    __bf16* Xv   = (__bf16*)(ws + 33554432LL);
    __bf16* WqT  = (__bf16*)(ws + 50331648LL);
    __bf16* WkT  = (__bf16*)(ws + 52428800LL);
    __bf16* WvT  = (__bf16*)(ws + 54525952LL);
    __bf16* WoT  = (__bf16*)(ws + 56623104LL);
    float*  projF= (float*) (ws + 58720256LL);                    // [8192][1024] f32
    __bf16* qh   = (__bf16*)(ws + 92274688LL);                    // [B][H][S][64]
    __bf16* kh   = (__bf16*)(ws + 109051904LL);
    __bf16* vTT  = (__bf16*)(ws + 125829120LL);                   // [B][H][64][S]
    __bf16* ctx  = (__bf16*)(ws + 142606336LL);                   // [8192][1024]

    const dim3 blk(256);
    const long long n4in = (long long)BSROWS * DMODEL / 4;        // 2097152

    // 1) convert activations to bf16
    f32_to_bf16_kernel<<<8192, blk, 0, stream>>>(Q, Xq, n4in);
    f32_to_bf16_kernel<<<8192, blk, 0, stream>>>(K, Xk, n4in);
    f32_to_bf16_kernel<<<8192, blk, 0, stream>>>(V, Xv, n4in);
    // 2) transpose+convert weights to [N][K] bf16
    transpose_w_kernel<<<1024, blk, 0, stream>>>(Wq, WqT);
    transpose_w_kernel<<<1024, blk, 0, stream>>>(Wk, WkT);
    transpose_w_kernel<<<1024, blk, 0, stream>>>(Wv, WvT);
    transpose_w_kernel<<<1024, blk, 0, stream>>>(Wo, WoT);

    // 3) projections + head packing (q gets 1/sqrt(64) folded in)
    dim3 gProj(DMODEL / 64, BSROWS / 128, 1);
    gemm_nt_bias_kernel<<<gProj, blk, 0, stream>>>(Xq, WqT, projF, bq, DMODEL);
    pack_heads_kernel<<<8192, blk, 0, stream>>>(projF, qh, 0.125f);
    gemm_nt_bias_kernel<<<gProj, blk, 0, stream>>>(Xk, WkT, projF, bk, DMODEL);
    pack_heads_kernel<<<8192, blk, 0, stream>>>(projF, kh, 1.0f);
    gemm_nt_bias_kernel<<<gProj, blk, 0, stream>>>(Xv, WvT, projF, bv, DMODEL);
    pack_vT_kernel<<<8192, blk, 0, stream>>>(projF, vTT);

    // 4) fused scores + softmax + context
    fused_attn_kernel<<<dim3(SEQ / 16, BATCH * NHEADS), dim3(512), 0, stream>>>(
        qh, kh, vTT, attnF, ctx);

    // 5) output projection: ctx [8192][1024] x Wo^T + bo -> outF
    gemm_nt_bias_kernel<<<gProj, blk, 0, stream>>>(ctx, WoT, outF, bo, DMODEL);
}
